// LSTM_9311489098195
// MI455X (gfx1250) — matrix operand order repaired and checked by reading, not compiled, not run on hardware
//
#include <hip/hip_runtime.h>
#include <hip/hip_bf16.h>

// ---------------------------------------------------------------------------
// LSTM on MI455X (gfx1250): persistent bf16-WMMA kernel.
//   Z^T = Wc * [h|x_t]^T  with Wc = [W_hh | W_ih]  (4H x 576), bf16
//   - Weights staged in LDS once (time-invariant): A operand fed by ds_load,
//     B operand (h ping-pong between WGs) fed by global_load -> the two
//     operand streams use independent memory pipes.
//   - Gates fused in registers (i,f,g,o accumulators share VGPR layout),
//     c kept in VGPRs for all 730 steps.
//   - One grid barrier per timestep; K-loop unroll capped to avoid spills.
// ---------------------------------------------------------------------------

typedef __attribute__((ext_vector_type(16))) __bf16 v16bf;
typedef __attribute__((ext_vector_type(8)))  __bf16 v8bf;
typedef __attribute__((ext_vector_type(8)))  float  v8f;

#define B_    1024
#define T_    730
#define IN_   64
#define H_    512
#define KTOT  (H_ + IN_)     // 576
#define NKB   (KTOT / 32)    // 18 K-blocks of 32
#define NKH   (H_ / 32)      // 16 h K-blocks
#define NBT   (B_ / 128)     // 8 batch tiles of 128
#define NHT   (H_ / 32)      // 16 h tiles of 32
#define GRID_ (NBT * NHT)    // 128 workgroups

// LDS: this WG's weight slice, fragment-ordered:
//   [gw = g*2 + wh][kb][lane][16]  -> 8 * 18 * 512 bf16 = 144 KB
#define LDS_ELEMS (8 * NKB * 512)
#define LDS_BYTES (LDS_ELEMS * 2)

__device__ __forceinline__ float fsig(float x)  { return 1.0f / (1.0f + __expf(-x)); }
__device__ __forceinline__ float ftanh_(float x){ return 1.0f - 2.0f / (__expf(2.0f * x) + 1.0f); }

// ---------------------------------------------------------------------------
// Pre-pass: swizzle combined weights [W_hh | W_ih] (f32) into bf16 stored in
// the exact per-lane order of the 16x32 bf16 WMMA A-fragment, so a fragment
// load is one contiguous 32-byte read per lane.
//   A 16x32 bf16 layout (contiguous, matches B): lane<16: row=lane,  K = {0..15}
//                                     lane>=16: row=lane-16, K = {16..31}
//   vector element e -> K_local = e + (lane>=16 ? 16:0)
// ---------------------------------------------------------------------------
__global__ void swizzle_w(const float* __restrict__ Wih,
                          const float* __restrict__ Whh,
                          __bf16* __restrict__ wsz)
{
    int idx = blockIdx.x * 256 + threadIdx.x;          // ((rb*NKB+kb)*32+lane)*16+e
    if (idx >= 128 * NKB * 512) return;
    int e    = idx & 15;
    int lane = (idx >> 4) & 31;
    int kb   = (idx >> 9) % NKB;
    int rb   = (idx >> 9) / NKB;                        // 16-row block in 4H=2048
    int row  = rb * 16 + (lane & 15);
    int kl   = e + ((lane >= 16) ? 16 : 0);
    int K    = kb * 32 + kl;
    float v  = (K < H_) ? Whh[(size_t)row * H_ + K]
                        : Wih[(size_t)row * IN_ + (K - H_)];
    wsz[idx] = (__bf16)v;
}

// ---------------------------------------------------------------------------
// Persistent recurrent kernel. 128 WGs x 256 threads (8 waves).
// Wave tile: 16 h-units x 32 batch, all 4 gates -> 8 f32 accum tiles.
// ---------------------------------------------------------------------------
__launch_bounds__(256, 1)
__global__ void lstm_persist(const float*  __restrict__ xd,     // [B,T,IN] f32
                             const float*  __restrict__ bias,   // [4H] f32
                             const __bf16* __restrict__ wsz,    // swizzled Wc
                             __bf16* __restrict__ h0,           // [B,H] ping
                             __bf16* __restrict__ h1,           // [B,H] pong
                             unsigned* __restrict__ counter)
{
    extern __shared__ __bf16 ldsW[];                  // 144 KB weight slice

    const int tid    = threadIdx.x;
    const int lane   = tid & 31;
    const int w      = tid >> 5;
    const int wh     = w & 1;                 // which 16-unit h block in WG
    const int wb     = w >> 1;                // which 32-batch group in WG
    const int bt     = blockIdx.x & (NBT - 1);
    const int ht     = blockIdx.x / NBT;
    const int hbase  = ht * 32 + wh * 16;     // wave's first h-unit
    const int bbase  = bt * 128 + wb * 32;    // wave's first batch row
    const int nlo    = lane & 15;             // B-operand / C-operand column
    const int hiHalf = (lane >= 16) ? 1 : 0;  // C rows +8, B/A K +16

    // ---- Prologue: stage this WG's weight slice into LDS (one wave per tile).
    // 144 tiles of 1 KB; tile gw = g*2+whl picks the 16-row block
    // g*512 + ht*32 + whl*16 of the 4H x 576 weight matrix.
    for (int tile = w; tile < 8 * NKB; tile += 8) {
        int kb  = tile % NKB;
        int gw  = tile / NKB;                  // g*2 + whl
        int g   = gw >> 1, whl = gw & 1;
        const __bf16* src = wsz + ((size_t)(g * (H_ / 16) + ht * 2 + whl) * NKB + kb) * 512
                                + (size_t)lane * 16;
        __bf16* dst = ldsW + (gw * NKB + kb) * 512 + lane * 16;
        *(uint4*)dst       = *(const uint4*)src;
        *(uint4*)(dst + 8) = *(const uint4*)(src + 8);
    }
    __syncthreads();

    // LDS base offsets of this wave's A-fragment streams per gate.
    int albase[4];
#pragma unroll
    for (int g = 0; g < 4; ++g)
        albase[g] = ((g * 2 + wh) * NKB) * 512 + lane * 16;

    // Bias for this thread's 8 accumulator rows per gate (C layout:
    // VGPR v, lane half -> row M = v + hiHalf*8).
    float bpre[4][8];
#pragma unroll
    for (int g = 0; g < 4; ++g)
#pragma unroll
        for (int v = 0; v < 8; ++v)
            bpre[g][v] = bias[g * H_ + hbase + v + hiHalf * 8];

    // Per-lane row pointers for the B operand (batch rows).
    const size_t brow0 = (size_t)(bbase + nlo) * H_;
    const size_t brow1 = (size_t)(bbase + 16 + nlo) * H_;
    const float* xrow0 = xd + (size_t)(bbase + nlo) * (T_ * IN_);
    const float* xrow1 = xd + (size_t)(bbase + 16 + nlo) * (T_ * IN_);
    const int    kofs  = hiHalf * 16;   // K offset for this lane half

    // Cell state lives in registers for the whole sequence.
    v8f cc[2];
#pragma unroll
    for (int nb = 0; nb < 2; ++nb)
#pragma unroll
        for (int v = 0; v < 8; ++v) cc[nb][v] = 0.0f;

    const unsigned nwg = gridDim.x;

    for (int t = 0; t < T_; ++t) {
        const __bf16* hr = (t & 1) ? h1 : h0;   // read buffer
        __bf16*       hw = (t & 1) ? h0 : h1;   // write buffer

        v8f acc[4][2];
#pragma unroll
        for (int g = 0; g < 4; ++g)
#pragma unroll
            for (int nb = 0; nb < 2; ++nb)
#pragma unroll
                for (int v = 0; v < 8; ++v) acc[g][nb][v] = bpre[g][v];

        // ---- h part of K (kb = 0..15): A from LDS, B from global ----
#pragma unroll 2
        for (int kb = 0; kb < NKH; ++kb) {
            const int k0 = kb * 32 + kofs;
            v16bf bf0 = *(const v16bf*)(hr + brow0 + k0);
            v16bf bf1 = *(const v16bf*)(hr + brow1 + k0);
#pragma unroll
            for (int g = 0; g < 4; ++g) {
                v16bf af = *(const v16bf*)(ldsW + albase[g] + kb * 512);
                acc[g][0] = __builtin_amdgcn_wmma_f32_16x16x32_bf16(
                    false, af, false, bf0, (short)0, acc[g][0], false, false);
                acc[g][1] = __builtin_amdgcn_wmma_f32_16x16x32_bf16(
                    false, af, false, bf1, (short)0, acc[g][1], false, false);
            }
        }

        // ---- x part of K (kb = 16..17): convert f32 -> bf16 on the fly ----
#pragma unroll 1
        for (int kb = NKH; kb < NKB; ++kb) {
            const int xo = (kb - NKH) * 32 + kofs;
            const float* p0 = xrow0 + (size_t)t * IN_ + xo;
            const float* p1 = xrow1 + (size_t)t * IN_ + xo;
            v16bf bf0, bf1;
#pragma unroll
            for (int e = 0; e < 16; ++e) { bf0[e] = (__bf16)p0[e]; bf1[e] = (__bf16)p1[e]; }
#pragma unroll
            for (int g = 0; g < 4; ++g) {
                v16bf af = *(const v16bf*)(ldsW + albase[g] + kb * 512);
                acc[g][0] = __builtin_amdgcn_wmma_f32_16x16x32_bf16(
                    false, af, false, bf0, (short)0, acc[g][0], false, false);
                acc[g][1] = __builtin_amdgcn_wmma_f32_16x16x32_bf16(
                    false, af, false, bf1, (short)0, acc[g][1], false, false);
            }
        }

        // Gate fusion in registers: all four gate accumulators share the
        // (VGPR,lane) -> (h-unit, batch) mapping.
#pragma unroll
        for (int nb = 0; nb < 2; ++nb) {
            v8bf hv;
#pragma unroll
            for (int v = 0; v < 8; ++v) {
                float ig = fsig(acc[0][nb][v]);
                float fg = fsig(acc[1][nb][v]);
                float gg = ftanh_(acc[2][nb][v]);
                float og = fsig(acc[3][nb][v]);
                float c  = fg * cc[nb][v] + ig * gg;
                cc[nb][v] = c;
                hv[v] = (__bf16)(og * ftanh_(c));
            }
            // rows j = v + hiHalf*8 are consecutive -> one 16B store.
            __bf16* dst = hw + (size_t)(bbase + nb * 16 + nlo) * H_ + hbase + hiHalf * 8;
            *(v8bf*)dst = hv;
        }

        // Grid-wide barrier (monotonic counter, release/acquire agent scope).
        __syncthreads();
        if (tid == 0) {
            __hip_atomic_fetch_add(counter, 1u, __ATOMIC_RELEASE, __HIP_MEMORY_SCOPE_AGENT);
            const unsigned target = (unsigned)(t + 1) * nwg;
            while (__hip_atomic_load(counter, __ATOMIC_ACQUIRE, __HIP_MEMORY_SCOPE_AGENT) < target)
                __builtin_amdgcn_s_sleep(2);
        }
        __syncthreads();
    }
}

// ---------------------------------------------------------------------------
// Head: out[b] = relu(dot(h_T[b,:], Wl) + bl). One wave per batch row.
// ---------------------------------------------------------------------------
__global__ void head_k(const __bf16* __restrict__ hfin,
                       const float*  __restrict__ Wl,
                       const float*  __restrict__ bl,
                       float* __restrict__ out)
{
    const int lane = threadIdx.x & 31;
    const int b    = blockIdx.x * 8 + (threadIdx.x >> 5);
    float s = 0.0f;
    for (int k = lane; k < H_; k += 32)
        s += (float)hfin[(size_t)b * H_ + k] * Wl[k];
#pragma unroll
    for (int off = 16; off; off >>= 1)
        s += __shfl_xor(s, off, 32);
    if (lane == 0) {
        float r = s + bl[0];
        out[b] = r > 0.0f ? r : 0.0f;
    }
}

// ---------------------------------------------------------------------------
extern "C" void kernel_launch(void* const* d_in, const int* in_sizes, int n_in,
                              void* d_out, int out_size, void* d_ws, size_t ws_size,
                              hipStream_t stream)
{
    (void)in_sizes; (void)n_in; (void)out_size; (void)ws_size;
    const float* xd   = (const float*)d_in[0];
    const float* Wih  = (const float*)d_in[1];
    const float* Whh  = (const float*)d_in[2];
    const float* bias = (const float*)d_in[3];
    const float* Wl   = (const float*)d_in[4];
    const float* bl   = (const float*)d_in[5];
    float* out = (float*)d_out;

    char* ws = (char*)d_ws;
    const size_t HB = (size_t)B_ * H_ * sizeof(__bf16);   // 1 MB per h buffer
    unsigned* counter = (unsigned*)ws;                    // [0,256)
    __bf16* h0  = (__bf16*)(ws + 256);
    __bf16* h1  = (__bf16*)(ws + 256 + HB);
    __bf16* wsz = (__bf16*)(ws + 256 + 2 * HB);           // 2.25 MB swizzled Wc

    // Zero counter + both h buffers every call (graph-capture safe, deterministic).
    hipMemsetAsync(d_ws, 0, 256 + 2 * HB, stream);

    const int swTotal = 128 * NKB * 512;
    swizzle_w<<<(swTotal + 255) / 256, 256, 0, stream>>>(Wih, Whh, wsz);
    lstm_persist<<<GRID_, 256, LDS_BYTES, stream>>>(xd, bias, wsz, h0, h1, counter);
    head_k<<<B_ / 8, 256, 0, stream>>>(h0, Wl, bl, out);
}
